// KronFTLinear_72911364817073
// MI455X (gfx1250) — matrix operand order repaired and checked by reading, not compiled, not run on hardware
//
#include <hip/hip_runtime.h>
#include <hip/hip_bf16.h>
#include <stdint.h>

// ---------------------------------------------------------------------------
// KronFTLinear for MI455X (gfx1250): 3 bf16 WMMA GEMMs (base + 2 kron-expanded
// delta weights) fused with per-row gate coefficients.  Sparse ifft2 evaluated
// directly via LDS cosine tables (only 1000 / 819 nonzero frequencies).
// GEMM staging uses GLOBAL_LOAD_ASYNC_TO_LDS_B128 + double buffering when the
// toolchain exposes the async builtins; otherwise falls back to sync staging.
// ---------------------------------------------------------------------------

#define B_   4
#define L_   2048
#define NROW (B_ * L_)      // 8192
#define D_   4096
#define O_   4096
#define P_   2
#define NFP_ 1000
#define INP_ 2048           // D/P
#define OUTP_ 2048          // O/P
#define NGATE_ 819          // P*D/10
#define SCALING_ 150.0f

typedef __bf16 v16bf __attribute__((ext_vector_type(16)));
typedef float  v8f   __attribute__((ext_vector_type(8)));
typedef int    vi4_t __attribute__((vector_size(16)));   // matches builtin V4i

// ---- CDNA5 async global->LDS path (ASYNCcnt) ------------------------------
#if defined(__has_builtin)
#if __has_builtin(__builtin_amdgcn_global_load_async_to_lds_b128)
#define HAVE_ASYNC_LDS 1
#define ASYNC_CP16(gp, lp, off)                                              \
    __builtin_amdgcn_global_load_async_to_lds_b128(                          \
        (__attribute__((address_space(1))) vi4_t*)(gp),                      \
        (__attribute__((address_space(3))) vi4_t*)(lp), (off), 0)
#endif
#endif

#if defined(HAVE_ASYNC_LDS)
#if __has_builtin(__builtin_amdgcn_s_wait_asynccnt)
#define WAIT_ASYNC0() __builtin_amdgcn_s_wait_asynccnt(0)
#else
#define WAIT_ASYNC0() asm volatile("s_wait_asynccnt 0" ::: "memory")
#endif
#endif

static __device__ __forceinline__ unsigned short f2bf(float f) {
    unsigned u = __float_as_uint(f);
    unsigned r = u + 0x7FFFu + ((u >> 16) & 1u);   // round-to-nearest-even
    return (unsigned short)(r >> 16);
}

// ---------------------------------------------------------------- convert f32 -> bf16
__global__ __launch_bounds__(256) void cvt_bf16_kernel(const float* __restrict__ src,
                                                       unsigned short* __restrict__ dst) {
    size_t i = ((size_t)blockIdx.x * 256 + threadIdx.x) * 4;
    float4 v = *(const float4*)(src + i);
    unsigned short o0 = f2bf(v.x), o1 = f2bf(v.y), o2 = f2bf(v.z), o3 = f2bf(v.w);
    uint2 packed;
    packed.x = (unsigned)o0 | ((unsigned)o1 << 16);
    packed.y = (unsigned)o2 | ((unsigned)o3 << 16);
    *(uint2*)(dst + i) = packed;
}

// ---------------------------------------------------------------- gate matrix (2 x 4096)
// gate_m[e,d] = (1/(P*D)) * sum_g gp[g] * cos(2*pi*(e*gr/2 + d*gc/4096))
__global__ __launch_bounds__(256) void build_gate_m_kernel(const float* __restrict__ gp,
                                                           const int* __restrict__ gidx,
                                                           float* __restrict__ gm) {
    __shared__ float sp[NGATE_];
    __shared__ int   sg[NGATE_];
    int tid = threadIdx.x;
    for (int t = tid; t < NGATE_; t += 256) { sp[t] = gp[t]; sg[t] = gidx[t]; }
    __syncthreads();
    int ed = blockIdx.x * 256 + tid;           // 8192 total
    int e = ed >> 12, d = ed & 4095;
    float a = 0.f;
    for (int k = 0; k < NGATE_; ++k) {
        int id = sg[k];
        int gr = id >> 12;                      // / 4096
        int gc = id & 4095;                     // % 4096
        int t = (e * gr * 2048 + d * gc) & 4095;
        a += sp[k] * __cosf((float)t * (6.283185307f / 4096.f));
    }
    gm[e * 4096 + d] = a * (1.0f / 8192.0f);
}

// ---------------------------------------------------------------- delta weights (bf16)
// A[m,n] = (1/(2048*2048)) * sum_k s_k * cos(2*pi*(m*r_k + n*c_k)/2048)
// Wd_p[2m+i, 2n+j] = 150 * A[m,n] * B_p[i,j]
__global__ __launch_bounds__(256) void build_delta_w_kernel(const float* __restrict__ spec,
                                                            const int* __restrict__ idx,
                                                            const float* __restrict__ Blist,
                                                            unsigned short* __restrict__ Wd0,
                                                            unsigned short* __restrict__ Wd1) {
    __shared__ float tbl[2048];
    __shared__ float ss[NFP_];
    __shared__ int   si[NFP_];
    int p = blockIdx.z;
    int tid = threadIdx.x;
    for (int t = tid; t < 2048; t += 256)
        tbl[t] = __cosf((float)t * (6.283185307f / 2048.f));
    for (int t = tid; t < NFP_; t += 256) { ss[t] = spec[p * NFP_ + t]; si[t] = idx[p * NFP_ + t]; }
    __syncthreads();
    int flat = blockIdx.x * 256 + tid;          // 2048*2048 per pack
    int m = flat >> 11, n = flat & 2047;
    float a = 0.f;
    #pragma unroll 4
    for (int k = 0; k < NFP_; ++k) {
        int id = si[k];
        int r = id >> 11;                        // / 2048
        int c = id & 2047;                       // % 2048
        a += ss[k] * tbl[(m * r + n * c) & 2047];
    }
    a *= (SCALING_ / (2048.0f * 2048.0f));
    unsigned short* W = p ? Wd1 : Wd0;
    const float* Bp = Blist + p * 4;
    size_t ro = (size_t)(2 * m) * D_ + 2 * n;
    W[ro]          = f2bf(a * Bp[0]);
    W[ro + 1]      = f2bf(a * Bp[1]);
    W[ro + D_]     = f2bf(a * Bp[2]);
    W[ro + D_ + 1] = f2bf(a * Bp[3]);
}

// ---------------------------------------------------------------- gate weights (softmax)
__global__ __launch_bounds__(256) void gate_softmax_kernel(const float* __restrict__ x,
                                                           const float* __restrict__ gm,
                                                           float* __restrict__ gate) {
    int n = blockIdx.x * 8 + (threadIdx.x >> 5);
    int lane = threadIdx.x & 31;
    const float* xr = x + (size_t)n * D_;
    float t0 = 0.f, t1 = 0.f;
    for (int d = lane; d < D_; d += 32) {
        float xv = xr[d];
        t0 += xv * gm[d];
        t1 += xv * gm[D_ + d];
    }
    for (int off = 16; off; off >>= 1) {
        t0 += __shfl_down(t0, off, 32);
        t1 += __shfl_down(t1, off, 32);
    }
    if (lane == 0) {
        float mx = fmaxf(t0, t1);
        float e0 = __expf(t0 - mx), e1 = __expf(t1 - mx);
        float s = 1.0f / (e0 + e1);
        gate[n * 2]     = e0 * s;
        gate[n * 2 + 1] = e1 * s;
    }
}

// ---------------------------------------------------------------- fused WMMA GEMM
// out[n,o] = bias[o] + X@W0^T + g0[n]*(X@W1^T) + g1[n]*(X@W2^T)
#define LDS_STRIDE 40   // 32 bf16 + 8 pad

__global__ __launch_bounds__(256) void fused_wmma_gemm_kernel(
    const unsigned short* __restrict__ X,     // NROW x D bf16
    const unsigned short* __restrict__ W0,
    const unsigned short* __restrict__ W1,
    const unsigned short* __restrict__ W2,
    const float* __restrict__ gate,           // NROW x 2
    const float* __restrict__ bias,           // O
    float* __restrict__ out) {                // NROW x O (f32)
    __shared__ unsigned short sX[2][128 * LDS_STRIDE];
    __shared__ unsigned short sW[2][128 * LDS_STRIDE];
    __shared__ float sG[256];

    const int tid  = threadIdx.x;
    const int lane = tid & 31;
    const int wid  = tid >> 5;
    const int bm = blockIdx.y * 128;
    const int bn = blockIdx.x * 128;

    // cache gate weights for this row block
    sG[tid] = gate[(size_t)(bm + (tid >> 1)) * 2 + (tid & 1)];

    const int wm = (wid >> 2) * 64;   // wave M offset: 0 / 64
    const int wn = (wid & 3) * 32;    // wave N offset: 0 / 32 / 64 / 96
    const int h   = lane >> 4;        // half-wave select
    const int l16 = lane & 15;

    v8f accF[4][2];
    #pragma unroll
    for (int mt = 0; mt < 4; ++mt)
        #pragma unroll
        for (int nt = 0; nt < 2; ++nt)
            #pragma unroll
            for (int v = 0; v < 8; ++v) accF[mt][nt][v] = 0.f;

    const unsigned short* Ws[3] = {W0, W1, W2};
    const int srow = tid >> 1;        // staging row 0..127
    const int shh  = tid & 1;         // staging half (16 bf16 each)
    const int soff = srow * LDS_STRIDE + shh * 16;

    for (int w = 0; w < 3; ++w) {
        const unsigned short* W = Ws[w];
        v8f acc[4][2];
        #pragma unroll
        for (int mt = 0; mt < 4; ++mt)
            #pragma unroll
            for (int nt = 0; nt < 2; ++nt)
                #pragma unroll
                for (int v = 0; v < 8; ++v) acc[mt][nt][v] = 0.f;

#if defined(HAVE_ASYNC_LDS)
        // ---- software pipeline: async global->LDS, double buffered ----
        {   // preload K tile 0 into buffer 0
            const unsigned short* gx = X + (size_t)(bm + srow) * D_ + shh * 16;
            const unsigned short* gw = W + (size_t)(bn + srow) * D_ + shh * 16;
            ASYNC_CP16(gx, &sX[0][soff], 0);
            ASYNC_CP16(gx, &sX[0][soff], 16);
            ASYNC_CP16(gw, &sW[0][soff], 0);
            ASYNC_CP16(gw, &sW[0][soff], 16);
            WAIT_ASYNC0();
            __syncthreads();
        }
        int parity = 0;
        for (int k = 0; k < D_; k += 32) {
            const int nxt = parity ^ 1;
            if (k + 32 < D_) {   // issue next tile while computing this one
                const unsigned short* gx = X + (size_t)(bm + srow) * D_ + (k + 32) + shh * 16;
                const unsigned short* gw = W + (size_t)(bn + srow) * D_ + (k + 32) + shh * 16;
                ASYNC_CP16(gx, &sX[nxt][soff], 0);
                ASYNC_CP16(gx, &sX[nxt][soff], 16);
                ASYNC_CP16(gw, &sW[nxt][soff], 0);
                ASYNC_CP16(gw, &sW[nxt][soff], 16);
            }
            // A fragments: 16x32 bf16, lane(h,m): K = 16h..16h+15 (contiguous)
            v16bf af[4];
            #pragma unroll
            for (int mt = 0; mt < 4; ++mt) {
                const unsigned short* base = &sX[parity][(wm + mt * 16 + l16) * LDS_STRIDE];
                ((uint4*)&af[mt])[0] = *(const uint4*)(base + 16 * h);
                ((uint4*)&af[mt])[1] = *(const uint4*)(base + 16 * h + 8);
            }
            // B fragments: 32x16 bf16, lane(h,n): K = 16h..16h+15 (contiguous)
            v16bf bfg[2];
            #pragma unroll
            for (int nt = 0; nt < 2; ++nt) {
                const unsigned short* base =
                    &sW[parity][(wn + nt * 16 + l16) * LDS_STRIDE + 16 * h];
                ((uint4*)&bfg[nt])[0] = *(const uint4*)(base);
                ((uint4*)&bfg[nt])[1] = *(const uint4*)(base + 8);
            }
            #pragma unroll
            for (int mt = 0; mt < 4; ++mt)
                #pragma unroll
                for (int nt = 0; nt < 2; ++nt)
                    acc[mt][nt] = __builtin_amdgcn_wmma_f32_16x16x32_bf16(
                        false, af[mt], false, bfg[nt], (short)0, acc[mt][nt], false, false);
            WAIT_ASYNC0();      // my async writes for the next tile are in LDS
            __syncthreads();    // everyone's writes visible; everyone's reads done
            parity = nxt;
        }
#else
        // ---- fallback: synchronous staging through VGPRs ----
        for (int k = 0; k < D_; k += 32) {
            __syncthreads();   // previous iter's fragment reads done
            {
                const uint4* gx = (const uint4*)(X + (size_t)(bm + srow) * D_ + k + shh * 16);
                const uint4* gw = (const uint4*)(W + (size_t)(bn + srow) * D_ + k + shh * 16);
                uint4 x0 = gx[0], x1 = gx[1];
                uint4 w0 = gw[0], w1 = gw[1];
                *(uint4*)(&sX[0][soff])     = x0;
                *(uint4*)(&sX[0][soff + 8]) = x1;
                *(uint4*)(&sW[0][soff])     = w0;
                *(uint4*)(&sW[0][soff + 8]) = w1;
                if (k + 32 < D_) {   // global_prefetch_b8 for next K tile
                    __builtin_prefetch(X + (size_t)(bm + srow) * D_ + k + 32, 0, 1);
                    __builtin_prefetch(W + (size_t)(bn + srow) * D_ + k + 32, 0, 1);
                }
            }
            __syncthreads();
            v16bf af[4];
            #pragma unroll
            for (int mt = 0; mt < 4; ++mt) {
                const unsigned short* base = &sX[0][(wm + mt * 16 + l16) * LDS_STRIDE];
                ((uint4*)&af[mt])[0] = *(const uint4*)(base + 16 * h);
                ((uint4*)&af[mt])[1] = *(const uint4*)(base + 16 * h + 8);
            }
            v16bf bfg[2];
            #pragma unroll
            for (int nt = 0; nt < 2; ++nt) {
                const unsigned short* base = &sW[0][(wn + nt * 16 + l16) * LDS_STRIDE + 16 * h];
                ((uint4*)&bfg[nt])[0] = *(const uint4*)(base);
                ((uint4*)&bfg[nt])[1] = *(const uint4*)(base + 8);
            }
            #pragma unroll
            for (int mt = 0; mt < 4; ++mt)
                #pragma unroll
                for (int nt = 0; nt < 2; ++nt)
                    acc[mt][nt] = __builtin_amdgcn_wmma_f32_16x16x32_bf16(
                        false, af[mt], false, bfg[nt], (short)0, acc[mt][nt], false, false);
        }
        __syncthreads();
#endif

        // merge with per-row coefficient (C layout: M = v + 8*(lane/16))
        #pragma unroll
        for (int mt = 0; mt < 4; ++mt)
            #pragma unroll
            for (int v = 0; v < 8; ++v) {
                int lrow = wm + mt * 16 + v + 8 * h;
                float c = (w == 0) ? 1.0f : sG[lrow * 2 + (w - 1)];
                #pragma unroll
                for (int nt = 0; nt < 2; ++nt)
                    accF[mt][nt][v] += c * acc[mt][nt][v];
            }
    }

    // epilogue: add bias and store f32
    #pragma unroll
    for (int nt = 0; nt < 2; ++nt) {
        int col = bn + wn + nt * 16 + l16;
        float bv = bias[col];
        #pragma unroll
        for (int mt = 0; mt < 4; ++mt)
            #pragma unroll
            for (int v = 0; v < 8; ++v) {
                int row = bm + wm + mt * 16 + v + 8 * h;
                out[(size_t)row * O_ + col] = accF[mt][nt][v] + bv;
            }
    }
}

// ---------------------------------------------------------------- host launch
extern "C" void kernel_launch(void* const* d_in, const int* in_sizes, int n_in,
                              void* d_out, int out_size, void* d_ws, size_t ws_size,
                              hipStream_t stream) {
    const float* x       = (const float*)d_in[0];   // (4,2048,4096)
    const float* base_w  = (const float*)d_in[1];   // (4096,4096)
    const float* base_b  = (const float*)d_in[2];   // (4096,)
    const float* spectrum= (const float*)d_in[3];   // (2,1000)
    const float* B_list  = (const float*)d_in[4];   // (2,2,2)
    const float* gate_p  = (const float*)d_in[5];   // (819,)
    const int*   idx     = (const int*)d_in[6];     // (2,1000)
    const int*   gidx    = (const int*)d_in[7];     // (819,)
    float* out = (float*)d_out;

    // workspace layout (requires ~160.1 MB)
    char* ws = (char*)d_ws;
    const size_t OFF_XBF = 0;                                   // 8192*4096 bf16 = 64 MB
    const size_t OFF_WB  = OFF_XBF + (size_t)NROW * D_ * 2;     // 4096*4096 bf16 = 32 MB
    const size_t OFF_WD0 = OFF_WB  + (size_t)O_ * D_ * 2;
    const size_t OFF_WD1 = OFF_WD0 + (size_t)O_ * D_ * 2;
    const size_t OFF_GM  = OFF_WD1 + (size_t)O_ * D_ * 2;       // 2*4096 f32
    const size_t OFF_GW  = OFF_GM  + (size_t)P_ * D_ * 4;       // 8192*2 f32
    unsigned short* Xbf = (unsigned short*)(ws + OFF_XBF);
    unsigned short* Wb  = (unsigned short*)(ws + OFF_WB);
    unsigned short* Wd0 = (unsigned short*)(ws + OFF_WD0);
    unsigned short* Wd1 = (unsigned short*)(ws + OFF_WD1);
    float* gm   = (float*)(ws + OFF_GM);
    float* gate = (float*)(ws + OFF_GW);

    // 1) bf16 conversions
    cvt_bf16_kernel<<<((size_t)NROW * D_) / 4 / 256, 256, 0, stream>>>(x, Xbf);
    cvt_bf16_kernel<<<((size_t)O_ * D_) / 4 / 256, 256, 0, stream>>>(base_w, Wb);
    // 2) gate matrix via sparse ifft2
    build_gate_m_kernel<<<(P_ * D_) / 256, 256, 0, stream>>>(gate_p, gidx, gm);
    // 3) delta weight matrices via sparse ifft2 + kron expansion (per pack)
    build_delta_w_kernel<<<dim3((OUTP_ * INP_) / 256, 1, P_), 256, 0, stream>>>(
        spectrum, idx, B_list, Wd0, Wd1);
    // 4) gate weights (softmax over packs)
    gate_softmax_kernel<<<NROW / 8, 256, 0, stream>>>(x, gm, gate);
    // 5) fused triple GEMM with gated combine + bias
    fused_wmma_gemm_kernel<<<dim3(O_ / 128, NROW / 128), 256, 0, stream>>>(
        Xbf, Wb, Wd0, Wd1, gate, base_b, out);
}
